// mamba_Attention_70076686402463
// MI455X (gfx1250) — matrix operand order repaired and checked by reading, not compiled, not run on hardware
//
#include <hip/hip_runtime.h>
#include <math.h>

typedef __attribute__((ext_vector_type(16))) __bf16 v16bf;
typedef __attribute__((ext_vector_type(8)))  __bf16 v8bf;
typedef __attribute__((ext_vector_type(8)))  float  v8f;

#define BATCH 2
#define LSEQ  1024
#define DMODEL 1024
#define ED    1024
#define NS    16
#define DTR   64
#define KCONV 4
#define TOK   (BATCH * LSEQ)        // 2048 tokens
#define DBCW  (DTR + 2 * NS)        // 96

__device__ __forceinline__ float siluf(float v) { return v / (1.0f + __expf(-v)); }

// ---------------- elementwise converts ----------------
__global__ void cvt_f32_bf16(const float* __restrict__ s, __bf16* __restrict__ d, int n) {
  int i = blockIdx.x * blockDim.x + threadIdx.x;
  if (i < n) d[i] = (__bf16)s[i];
}

__global__ void cvt2d_f32_bf16(const float* __restrict__ s, __bf16* __restrict__ d,
                               int rows, int cols, int sstride) {
  int i = blockIdx.x * blockDim.x + threadIdx.x;
  if (i >= rows * cols) return;
  int r = i / cols, c = i - r * cols;
  d[i] = (__bf16)s[(size_t)r * sstride + c];
}

// ---------------- WMMA GEMM: C[T x N] = A[T x K](bf16) * W[N x K](bf16)^T ----------------
// One wave computes a 16x32 output tile with two v_wmma_f32_16x16x32_bf16 accumulators.
__global__ void gemm_bf16_wmma(const __bf16* __restrict__ A, const __bf16* __restrict__ W,
                               float* __restrict__ C, int T, int N, int K) {
  const int lane = threadIdx.x & 31;
  const int wave = (blockIdx.x * blockDim.x + threadIdx.x) >> 5;
  const int nStrips = N >> 5;
  const int nTiles  = (T >> 4) * nStrips;
  if (wave >= nTiles) return;                 // wave-uniform: EXEC stays all-ones for WMMA
  const int m0   = (wave / nStrips) << 4;
  const int n0   = (wave % nStrips) << 5;
  const int half = (lane >> 4) & 1;
  const int l16  = lane & 15;

  // A: lane holds row m0+l16; K chunks depend on lane-half (contiguous: k = kk + 16*half + s, same order as B)
  const __bf16* Arow  = A + (size_t)(m0 + l16) * K;
  // B (=W^T): lane holds W row (n0 [+16] + l16), 16 contiguous bf16 per step
  const __bf16* Brow0 = W + (size_t)(n0 + l16) * K + 16 * half;
  const __bf16* Brow1 = W + (size_t)(n0 + 16 + l16) * K + 16 * half;

  v8f acc0 = {}; v8f acc1 = {};
  for (int kk = 0; kk < K; kk += 32) {
    union { v16bf v; v8bf h[2]; } a;
    a.h[0] = *(const v8bf*)(Arow + kk + 16 * half);       // K = kk + 16*half .. +7
    a.h[1] = *(const v8bf*)(Arow + kk + 16 * half + 8);   // K = kk + 16*half + 8 .. +15
    v16bf b0 = *(const v16bf*)(Brow0 + kk);               // K = kk + 16*half .. +15
    v16bf b1 = *(const v16bf*)(Brow1 + kk);
    acc0 = __builtin_amdgcn_wmma_f32_16x16x32_bf16(false, a.v, false, b0, (short)0, acc0, false, false);
    acc1 = __builtin_amdgcn_wmma_f32_16x16x32_bf16(false, a.v, false, b1, (short)0, acc1, false, false);
  }
  // C/D layout: VGPR r, lanes 0-15 -> M=r, lanes 16-31 -> M=8+r; col = lane&15
  float* Cb = C + (size_t)(m0 + half * 8) * N + n0 + l16;
#pragma unroll
  for (int r = 0; r < 8; ++r) {
    Cb[(size_t)r * N]      = acc0[r];
    Cb[(size_t)r * N + 16] = acc1[r];
  }
}

// ---------------- causal depthwise conv1d (K=4) + SiLU ----------------
__global__ void conv_silu(const float* __restrict__ xz, const float* __restrict__ conv_w,
                          const float* __restrict__ conv_b,
                          float* __restrict__ xi_f, __bf16* __restrict__ xi_h) {
  int i = blockIdx.x * blockDim.x + threadIdx.x;   // over TOK*ED
  if (i >= TOK * ED) return;
  int e = i & (ED - 1);
  int t = i >> 10;                                  // token index (ED == 1024)
  int l = t & (LSEQ - 1);
  int b = t >> 10;
  float acc = conv_b[e];
#pragma unroll
  for (int k = 0; k < KCONV; ++k) {
    int lk = l + k - (KCONV - 1);
    if (lk >= 0)
      acc += xz[(size_t)(b * LSEQ + lk) * (2 * ED) + e] * conv_w[e * KCONV + k];
  }
  float s = siluf(acc);
  xi_f[i] = s;
  xi_h[i] = (__bf16)s;
}

// ---------------- selective scan: one lane per (b, e, n) state chain ----------------
// 16 lanes (n=0..15) share (b,e); einsum over n via intra-wave shuffle reduce.
__global__ void selective_scan(const float* __restrict__ delta_pre, const float* __restrict__ dt_b,
                               const float* __restrict__ xi_f, const float* __restrict__ dbc,
                               const float* __restrict__ xz, const float* __restrict__ A_log,
                               const float* __restrict__ D_param, __bf16* __restrict__ y_h) {
  int tid = blockIdx.x * blockDim.x + threadIdx.x;   // 0 .. BATCH*ED*NS-1 = 32767
  int n = tid & (NS - 1);
  int e = (tid >> 4) & (ED - 1);
  int b = tid >> 14;
  float Aen = -__expf(A_log[e * NS + n]);
  float dtb = dt_b[e];
  float Dp  = D_param[e];
  float h = 0.0f;
  for (int l = 0; l < LSEQ; ++l) {
    int t = b * LSEQ + l;
    float dpre  = delta_pre[(size_t)t * ED + e] + dtb;
    float delta = (dpre > 20.0f) ? dpre : log1pf(__expf(dpre));
    float xiv = xi_f[(size_t)t * ED + e];
    float Bv  = dbc[t * DBCW + DTR + n];
    float Cv  = dbc[t * DBCW + DTR + NS + n];
    h = __expf(delta * Aen) * h + (delta * xiv) * Bv;
    float yv = h * Cv;
#pragma unroll
    for (int off = 1; off < NS; off <<= 1) yv += __shfl_xor(yv, off, NS);
    if (n == 0) {
      float z = xz[(size_t)t * (2 * ED) + ED + e];
      y_h[(size_t)t * ED + e] = (__bf16)((yv + Dp * xiv) * siluf(z));
    }
  }
}

// ---------------- host orchestration ----------------
static inline int ceil_div(int a, int b) { return (a + b - 1) / b; }

extern "C" void kernel_launch(void* const* d_in, const int* in_sizes, int n_in,
                              void* d_out, int out_size, void* d_ws, size_t ws_size,
                              hipStream_t stream) {
  const float* x          = (const float*)d_in[0];
  const float* in_proj_w  = (const float*)d_in[1];
  const float* conv_w     = (const float*)d_in[2];
  const float* conv_b     = (const float*)d_in[3];
  const float* x_proj_w   = (const float*)d_in[4];
  const float* dt_proj_w  = (const float*)d_in[5];
  const float* dt_proj_b  = (const float*)d_in[6];
  const float* A_log      = (const float*)d_in[7];
  const float* D_param    = (const float*)d_in[8];
  const float* out_proj_w = (const float*)d_in[9];
  float* out = (float*)d_out;

  char* p = (char*)d_ws;
  auto alloc = [&](size_t bytes) -> char* {
    char* r = p; p += (bytes + 255) & ~(size_t)255; return r;
  };
  __bf16* x_h     = (__bf16*)alloc((size_t)TOK * DMODEL * 2);
  __bf16* w_in_h  = (__bf16*)alloc((size_t)2 * ED * DMODEL * 2);
  __bf16* w_xp_h  = (__bf16*)alloc((size_t)DBCW * ED * 2);
  __bf16* w_dt_h  = (__bf16*)alloc((size_t)ED * DTR * 2);
  __bf16* w_out_h = (__bf16*)alloc((size_t)DMODEL * ED * 2);
  float*  xz      = (float*) alloc((size_t)TOK * 2 * ED * 4);
  float*  xi_f    = (float*) alloc((size_t)TOK * ED * 4);
  __bf16* xi_h    = (__bf16*)alloc((size_t)TOK * ED * 2);
  float*  dbc     = (float*) alloc((size_t)TOK * DBCW * 4);
  __bf16* dbc_h   = (__bf16*)alloc((size_t)TOK * DTR * 2);
  float*  dpre    = (float*) alloc((size_t)TOK * ED * 4);
  __bf16* y_h     = (__bf16*)alloc((size_t)TOK * ED * 2);

  const int BS = 256;
  // 1) fp32 -> bf16 operand staging (weights + activations)
  cvt_f32_bf16<<<ceil_div(TOK * DMODEL, BS), BS, 0, stream>>>(x, x_h, TOK * DMODEL);
  cvt_f32_bf16<<<ceil_div(2 * ED * DMODEL, BS), BS, 0, stream>>>(in_proj_w, w_in_h, 2 * ED * DMODEL);
  cvt_f32_bf16<<<ceil_div(DBCW * ED, BS), BS, 0, stream>>>(x_proj_w, w_xp_h, DBCW * ED);
  cvt_f32_bf16<<<ceil_div(ED * DTR, BS), BS, 0, stream>>>(dt_proj_w, w_dt_h, ED * DTR);
  cvt_f32_bf16<<<ceil_div(DMODEL * ED, BS), BS, 0, stream>>>(out_proj_w, w_out_h, DMODEL * ED);

  // 2) xz = x @ in_proj_w^T   (2048 x 2048 x 1024)
  {
    int waves = (TOK / 16) * ((2 * ED) / 32);
    gemm_bf16_wmma<<<ceil_div(waves * 32, BS), BS, 0, stream>>>(x_h, w_in_h, xz, TOK, 2 * ED, DMODEL);
  }
  // 3) depthwise conv + SiLU
  conv_silu<<<ceil_div(TOK * ED, BS), BS, 0, stream>>>(xz, conv_w, conv_b, xi_f, xi_h);

  // 4) dbc = xi @ x_proj_w^T  (2048 x 96 x 1024)
  {
    int waves = (TOK / 16) * (DBCW / 32);
    gemm_bf16_wmma<<<ceil_div(waves * 32, BS), BS, 0, stream>>>(xi_h, w_xp_h, dbc, TOK, DBCW, ED);
  }
  // 5) bf16 stage of dbc[:, :64], then delta_pre = dbc_dt @ dt_proj_w^T  (2048 x 1024 x 64)
  cvt2d_f32_bf16<<<ceil_div(TOK * DTR, BS), BS, 0, stream>>>(dbc, dbc_h, TOK, DTR, DBCW);
  {
    int waves = (TOK / 16) * (ED / 32);
    gemm_bf16_wmma<<<ceil_div(waves * 32, BS), BS, 0, stream>>>(dbc_h, w_dt_h, dpre, TOK, ED, DTR);
  }
  // 6) selective scan (32K independent chains), fused softplus / gating / SiLU(z)
  selective_scan<<<ceil_div(BATCH * ED * NS, BS), BS, 0, stream>>>(
      dpre, dt_proj_b, xi_f, dbc, xz, A_log, D_param, y_h);

  // 7) out = y @ out_proj_w^T (2048 x 1024 x 1024), fp32 result straight to d_out
  {
    int waves = (TOK / 16) * (DMODEL / 32);
    gemm_bf16_wmma<<<ceil_div(waves * 32, BS), BS, 0, stream>>>(y_h, w_out_h, out, TOK, DMODEL, ED);
  }
  (void)in_sizes; (void)n_in; (void)out_size; (void)ws_size;
}
